// GraphSAGETripletRankingLoss_57801669869913
// MI455X (gfx1250) — matrix operand order repaired and checked by reading, not compiled, not run on hardware
//
#include <hip/hip_runtime.h>

typedef __attribute__((ext_vector_type(16))) _Float16 v16h;
typedef __attribute__((ext_vector_type(8)))  _Float16 h8;
typedef __attribute__((ext_vector_type(4)))  _Float16 h4;
typedef __attribute__((ext_vector_type(8)))  float    v8f;

#define IN_C 128
#define HID  128
#define OUT_C 64

// ---------- small utility kernels ----------

__global__ void cast_f32_f16(const float* __restrict__ in, _Float16* __restrict__ out, int n) {
    int i = blockIdx.x * blockDim.x + threadIdx.x;
    if (i < n) out[i] = (_Float16)in[i];
}

// W is [K][Nout] row-major f32; WT is [Nout][K] row-major f16
__global__ void transpose_w(const float* __restrict__ W, _Float16* __restrict__ WT, int K, int Nout) {
    int i = blockIdx.x * blockDim.x + threadIdx.x;
    if (i >= K * Nout) return;
    int n = i / K, k = i - n * K;
    WT[i] = (_Float16)W[k * Nout + n];
}

// ---------- mean aggregation (scatter-add) ----------
// one wave per edge; 32 lanes x 4 channels = 128 channels
__global__ void agg_kernel(const _Float16* __restrict__ feat,
                           const int* __restrict__ src,
                           const int* __restrict__ dst,
                           float* __restrict__ agg,
                           float* __restrict__ cnt,
                           int E, int addCnt) {
    int edge = (blockIdx.x * blockDim.x + threadIdx.x) >> 5;
    int lane = threadIdx.x & 31;
    if (edge >= E) return;
    int s = src[edge], d = dst[edge];
    const _Float16* fp = feat + (long long)s * 128 + lane * 4;
    float* ap = agg + (long long)d * 128 + lane * 4;
    h4 v = *(const h4*)fp;
#pragma unroll
    for (int j = 0; j < 4; j++) atomicAdd(ap + j, (float)v[j]);
    if (addCnt && lane == 0) atomicAdd(&cnt[d], 1.0f);
}

__global__ void finalize_agg(const float* __restrict__ agg, const float* __restrict__ cnt,
                             _Float16* __restrict__ out, int total) {
    int i = blockIdx.x * blockDim.x + threadIdx.x;
    if (i >= total) return;
    int node = i >> 7;
    float c = fmaxf(cnt[node], 1.0f);
    out[i] = (_Float16)(agg[i] / c);
}

// ---------- fused dual-GEMM: D = relu?(A1@B1 + A2@B2 + bias) ----------
// A1,A2: [Nrows][128] f16 row-major. B1T,B2T: [NOUT][128] f16 (transposed weights).
// One wave computes a 16x16 tile; blockDim.x/32 waves cover NOUT/16 column tiles.
__device__ inline v16h mk16(h8 lo, h8 hi) {
    v16h v;
#pragma unroll
    for (int i = 0; i < 8; i++) { v[i] = lo[i]; v[i + 8] = hi[i]; }
    return v;
}

template <int NOUT, bool RELU, typename OUT_T>
__global__ void gemm_dual(const _Float16* __restrict__ A1,
                          const _Float16* __restrict__ A2,
                          const _Float16* __restrict__ B1T,
                          const _Float16* __restrict__ B2T,
                          const float* __restrict__ bias,
                          OUT_T* __restrict__ out, int Nrows) {
    const int K = 128;
    int wave = threadIdx.x >> 5;
    int lane = threadIdx.x & 31;
    int row0 = blockIdx.x * 16;
    int col0 = wave * 16;
    int r  = lane & 15;
    int hi = lane >> 4;            // 0 for lanes 0-15, 1 for lanes 16-31

    int arow = row0 + r;
    if (arow >= Nrows) arow = Nrows - 1;     // clamp (N is a multiple of 16 anyway)
    const _Float16* a1p = A1 + (long long)arow * K + hi * 8;   // A frag: K k0+ksel..+8 and +16..+24
    const _Float16* a2p = A2 + (long long)arow * K + hi * 8;
    int bcol = col0 + r;
    const _Float16* b1p = B1T + (long long)bcol * K + hi * 8;  // B frag: K k0+ksel..+8 and +16..+24
    const _Float16* b2p = B2T + (long long)bcol * K + hi * 8;

    v8f acc = {};
#pragma unroll
    for (int k0 = 0; k0 < K; k0 += 32) {
        v16h a1 = mk16(*(const h8*)(a1p + k0), *(const h8*)(a1p + k0 + 16));
        v16h b1 = mk16(*(const h8*)(b1p + k0), *(const h8*)(b1p + k0 + 16));
        acc = __builtin_amdgcn_wmma_f32_16x16x32_f16(false, a1, false, b1,
                                                     (short)0, acc, false, false);
        v16h a2 = mk16(*(const h8*)(a2p + k0), *(const h8*)(a2p + k0 + 16));
        v16h b2 = mk16(*(const h8*)(b2p + k0), *(const h8*)(b2p + k0 + 16));
        acc = __builtin_amdgcn_wmma_f32_16x16x32_f16(false, a2, false, b2,
                                                     (short)0, acc, false, false);
    }

    float bv = bias[col0 + r];
    int crow = row0 + hi * 8;
#pragma unroll
    for (int v = 0; v < 8; v++) {
        int rr = crow + v;
        if (rr < Nrows) {
            float val = acc[v] + bv;
            if (RELU) val = fmaxf(val, 0.0f);
            out[(long long)rr * NOUT + col0 + r] = (OUT_T)val;
        }
    }
}

// ---------- PairNorm statistics: column sums + total sum of squares ----------
__global__ void colstat(const float* __restrict__ h, int N,
                        float* __restrict__ colsum, float* __restrict__ totalss) {
    __shared__ float ssum[256];
    __shared__ float sss[256];
    int tid = threadIdx.x;
    int col = tid & 63;
    int g   = tid >> 6;     // 0..3
    float s = 0.f, q = 0.f;
    for (int row = blockIdx.x * 4 + g; row < N; row += gridDim.x * 4) {
        float v = h[(long long)row * 64 + col];
        s += v; q += v * v;
    }
    ssum[tid] = s; sss[tid] = q;
    __syncthreads();
    if (tid < 64) {
        float t = ssum[tid] + ssum[tid + 64] + ssum[tid + 128] + ssum[tid + 192];
        atomicAdd(colsum + col, t);
    }
    for (int off = 128; off > 0; off >>= 1) {
        if (tid < off) sss[tid] += sss[tid + off];
        __syncthreads();
    }
    if (tid == 0) atomicAdd(totalss, sss[0]);
}

__global__ void finalize_norm(const float* __restrict__ colsum, const float* __restrict__ totalss,
                              float* __restrict__ mean, float* __restrict__ invd, int N) {
    __shared__ float smu2[64];
    int c = threadIdx.x;   // 64 threads
    float mu = colsum[c] / (float)N;
    mean[c] = mu;
    smu2[c] = mu * mu;
    __syncthreads();
    for (int off = 32; off > 0; off >>= 1) {
        if (c < off) smu2[c] += smu2[c + off];
        __syncthreads();
    }
    if (c == 0) {
        float ss = (*totalss - (float)N * smu2[0]) / (float)N;
        *invd = rsqrtf(1e-6f + ss);
    }
}

__global__ void pn_apply(const float* __restrict__ h, const float* __restrict__ mean,
                         const float* __restrict__ invd, float* __restrict__ out, long long total) {
    long long i = (long long)blockIdx.x * blockDim.x + threadIdx.x;
    if (i >= total) return;
    int c = (int)(i & 63);
    out[i] = (h[i] - mean[c]) * (*invd);
}

// ---------- triplet margin loss ----------
__global__ void loss_kernel(const float* __restrict__ hn,
                            const int* __restrict__ psrc, const int* __restrict__ pdst,
                            const int* __restrict__ ndst, int E, float invE,
                            float* __restrict__ out) {
    __shared__ float sred[256];
    float acc = 0.f;
    for (int e = blockIdx.x * blockDim.x + threadIdx.x; e < E; e += gridDim.x * blockDim.x) {
        const float4* a = (const float4*)(hn + (long long)psrc[e] * 64);
        const float4* p = (const float4*)(hn + (long long)pdst[e] * 64);
        const float4* n = (const float4*)(hn + (long long)ndst[e] * 64);
        float pd = 0.f, nd = 0.f;
#pragma unroll
        for (int c = 0; c < 16; c++) {
            float4 av = a[c], pv = p[c], nv = n[c];
            float d0 = av.x - pv.x, d1 = av.y - pv.y, d2 = av.z - pv.z, d3 = av.w - pv.w;
            pd += d0 * d0 + d1 * d1 + d2 * d2 + d3 * d3;
            float e0 = av.x - nv.x, e1 = av.y - nv.y, e2 = av.z - nv.z, e3 = av.w - nv.w;
            nd += e0 * e0 + e1 * e1 + e2 * e2 + e3 * e3;
        }
        acc += fmaxf(pd - nd + 1.0f, 0.0f);
    }
    sred[threadIdx.x] = acc;
    __syncthreads();
    for (int off = 128; off > 0; off >>= 1) {
        if (threadIdx.x < off) sred[threadIdx.x] += sred[threadIdx.x + off];
        __syncthreads();
    }
    if (threadIdx.x == 0) atomicAdd(out, sred[0] * invE);
}

// ---------- host side ----------
extern "C" void kernel_launch(void* const* d_in, const int* in_sizes, int n_in,
                              void* d_out, int out_size, void* d_ws, size_t ws_size,
                              hipStream_t stream) {
    const float* x   = (const float*)d_in[0];
    const int*   pei = (const int*)d_in[1];   // [2][E]: src then dst
    const int*   nei = (const int*)d_in[2];
    const float* W1l = (const float*)d_in[3];
    const float* b1  = (const float*)d_in[4];
    const float* W1r = (const float*)d_in[5];
    const float* W2l = (const float*)d_in[6];
    const float* b2  = (const float*)d_in[7];
    const float* W2r = (const float*)d_in[8];

    const int N = in_sizes[0] / IN_C;
    const int E = in_sizes[1] / 2;

    // carve workspace (256B-aligned regions)
    char* ws = (char*)d_ws;
    auto carve = [&](size_t bytes) -> char* {
        char* p = ws;
        ws += (bytes + 255) & ~(size_t)255;
        return p;
    };
    _Float16* x_h   = (_Float16*)carve((size_t)N * 128 * 2);
    _Float16* h1_h  = (_Float16*)carve((size_t)N * 128 * 2);
    float*    aggf  = (float*)   carve((size_t)N * 128 * 4);
    float*    cnt   = (float*)   carve((size_t)N * 4);
    _Float16* agg_h = (_Float16*)carve((size_t)N * 128 * 2);
    float*    h2    = (float*)   carve((size_t)N * 64 * 4);
    float*    hn    = (float*)   carve((size_t)N * 64 * 4);
    _Float16* w1lT  = (_Float16*)carve(128 * 128 * 2);
    _Float16* w1rT  = (_Float16*)carve(128 * 128 * 2);
    _Float16* w2lT  = (_Float16*)carve(64 * 128 * 2);
    _Float16* w2rT  = (_Float16*)carve(64 * 128 * 2);
    float*    colsum  = (float*)carve(64 * 4);
    float*    totalss = (float*)carve(4);
    float*    meanb   = (float*)carve(64 * 4);
    float*    invd    = (float*)carve(4);

    hipMemsetAsync(aggf, 0, (size_t)N * 128 * 4, stream);
    hipMemsetAsync(cnt, 0, (size_t)N * 4, stream);
    hipMemsetAsync(colsum, 0, 64 * 4, stream);
    hipMemsetAsync(totalss, 0, 4, stream);
    hipMemsetAsync(d_out, 0, 4, stream);

    // precision conversion + weight transposes
    cast_f32_f16<<<(N * 128 + 255) / 256, 256, 0, stream>>>(x, x_h, N * 128);
    transpose_w<<<(128 * 128 + 255) / 256, 256, 0, stream>>>(W1l, w1lT, 128, 128);
    transpose_w<<<(128 * 128 + 255) / 256, 256, 0, stream>>>(W1r, w1rT, 128, 128);
    transpose_w<<<(128 * 64 + 255) / 256, 256, 0, stream>>>(W2l, w2lT, 128, 64);
    transpose_w<<<(128 * 64 + 255) / 256, 256, 0, stream>>>(W2r, w2rT, 128, 64);

    int aggBlocks = (E * 32 + 255) / 256;
    int rowTiles  = (N + 15) / 16;

    // layer 1: mean-agg + fused dual GEMM (relu)
    agg_kernel<<<aggBlocks, 256, 0, stream>>>(x_h, pei, pei + E, aggf, cnt, E, 1);
    finalize_agg<<<(N * 128 + 255) / 256, 256, 0, stream>>>(aggf, cnt, agg_h, N * 128);
    gemm_dual<HID, true, _Float16><<<rowTiles, 256, 0, stream>>>(agg_h, x_h, w1lT, w1rT, b1, h1_h, N);

    // layer 2: re-zero agg, same edge set (cnt reused)
    hipMemsetAsync(aggf, 0, (size_t)N * 128 * 4, stream);
    agg_kernel<<<aggBlocks, 256, 0, stream>>>(h1_h, pei, pei + E, aggf, cnt, E, 0);
    finalize_agg<<<(N * 128 + 255) / 256, 256, 0, stream>>>(aggf, cnt, agg_h, N * 128);
    gemm_dual<OUT_C, false, float><<<rowTiles, 128, 0, stream>>>(agg_h, h1_h, w2lT, w2rT, b2, h2, N);

    // PairNorm
    colstat<<<512, 256, 0, stream>>>(h2, N, colsum, totalss);
    finalize_norm<<<1, 64, 0, stream>>>(colsum, totalss, meanb, invd, N);
    pn_apply<<<(int)(((long long)N * 64 + 255) / 256), 256, 0, stream>>>(h2, meanb, invd, hn, (long long)N * 64);

    // triplet margin loss -> d_out[0]
    loss_kernel<<<2048, 256, 0, stream>>>(hn, pei, pei + E, nei + E, E, 1.0f / (float)E, (float*)d_out);
}
